// TargetAttention_55628416418169
// MI455X (gfx1250) — matrix operand order repaired and checked by reading, not compiled, not run on hardware
//
#include <hip/hip_runtime.h>

#define DEV __device__ __forceinline__

typedef __attribute__((ext_vector_type(16))) __bf16 v16bf;
typedef __attribute__((ext_vector_type(8)))  __bf16 v8bf;
typedef __attribute__((ext_vector_type(8)))  float  v8f;

constexpr int cB = 4;          // batch
constexpr int cD = 512;        // embed dim
constexpr int cL = 4096;       // seq len
constexpr int cN = 8921;       // n_labels
constexpr int cNT = (cN + 15) / 16;   // 558 label tiles
constexpr int cNpad = cNT * 16;       // 8928 (zero-padded W rows)

// ---- load 16 bf16 (two contiguous 16B runs) into a WMMA fragment ----
DEV v16bf ld16bf(const __bf16* p0, const __bf16* p1) {
  v8bf lo = *(const v8bf*)p0;
  v8bf hi = *(const v8bf*)p1;
  v16bf r;
#pragma unroll
  for (int i = 0; i < 8; ++i) { r[i] = lo[i]; r[i + 8] = hi[i]; }
  return r;
}

// ============================ prep kernels ============================

// W [N,D] fp32 -> Wbf [Npad,D] bf16, zero-padded rows
__global__ void ta_prep_w(const float* __restrict__ W, __bf16* __restrict__ Wbf) {
  size_t i = (size_t)blockIdx.x * 256 + threadIdx.x;
  if (i >= (size_t)cNpad * cD) return;
  int n = (int)(i / cD);
  Wbf[i] = (n < cN) ? (__bf16)W[i] : (__bf16)0.0f;
}

// V [B,D,L] fp32 -> Vbf same layout bf16 (rows contiguous in l = GEMM2 K-dim)
__global__ void ta_prep_v(const float* __restrict__ V, __bf16* __restrict__ Vbf) {
  size_t i = (size_t)blockIdx.x * 256 + threadIdx.x;
  if (i >= (size_t)cB * cD * cL) return;
  Vbf[i] = (__bf16)V[i];
}

// K [B,D,L] fp32 -> Kt [B,L,D] bf16 (tiled transpose; rows contiguous in d)
__global__ void ta_prep_kt(const float* __restrict__ K, __bf16* __restrict__ Kt) {
  __shared__ float tile[32][33];
  const int b  = blockIdx.z;
  const int l0 = blockIdx.x * 32;
  const int d0 = blockIdx.y * 32;
  const int tx = threadIdx.x;   // 0..31
  const int ty = threadIdx.y;   // 0..7
  const float* src = K + ((size_t)b * cD + d0) * cL + l0;
#pragma unroll
  for (int k = 0; k < 4; ++k)
    tile[ty + 8 * k][tx] = src[(size_t)(ty + 8 * k) * cL + tx];
  __syncthreads();
  __bf16* dst = Kt + ((size_t)b * cL + l0) * cD + d0;
#pragma unroll
  for (int k = 0; k < 4; ++k)
    dst[(size_t)(ty + 8 * k) * cD + tx] = (__bf16)tile[tx][ty + 8 * k];
}

// ============================ main kernel =============================
__global__ __launch_bounds__(128) void ta_attn(
    const __bf16* __restrict__ Wbf, const __bf16* __restrict__ Kt,
    const __bf16* __restrict__ Vbf, float* __restrict__ gA,
    float* __restrict__ gC) {
  __shared__ float sPart[4][16];
  __shared__ float sMax[16];
  __shared__ float sInv[16];

  const int tile = blockIdx.x;
  const int b    = tile / cNT;
  const int n0   = (tile % cNT) * 16;
  const int tid  = threadIdx.x;
  const int wv   = tid >> 5;
  const int lane = tid & 31;
  const int hf   = lane >> 4;       // 0: lanes 0-15, 1: lanes 16-31
  const int lq   = lane & 15;
  const int kbA  = hf * 8;          // A-frag K-run base (16-bit A layout)
  const int kbB  = hf * 16;         // B-frag K base    (contiguous k = 16*hf + s)

  // ------- Phase 1: E = W@K, two interleaved l-blocks per iteration -------
  float rmax[8];
#pragma unroll
  for (int r = 0; r < 8; ++r) rmax[r] = -3.0e38f;

  const __bf16* Wrow = Wbf + (size_t)(n0 + lq) * cD;   // A row m = lane&15

  for (int p = wv; p < cL / 32; p += 4) {
    const int l0 = p * 32;                             // blocks l0 and l0+16
    v8f c0 = {0.f, 0.f, 0.f, 0.f, 0.f, 0.f, 0.f, 0.f};
    v8f c1 = {0.f, 0.f, 0.f, 0.f, 0.f, 0.f, 0.f, 0.f};
    const __bf16* Kr0 = Kt + ((size_t)b * cL + l0 + lq) * cD;
    const __bf16* Kr1 = Kr0 + (size_t)16 * cD;
    // group-of-4 K-steps: issue 8 B-fragment loads, then 8 WMMAs
#pragma unroll
    for (int g = 0; g < cD; g += 128) {
      v16bf bf0[4], bf1[4];
#pragma unroll
      for (int j = 0; j < 4; ++j) {
        const int dk = g + 32 * j;
        bf0[j] = ld16bf(Kr0 + dk + kbB, Kr0 + dk + kbB + 8);
        bf1[j] = ld16bf(Kr1 + dk + kbB, Kr1 + dk + kbB + 8);
      }
#pragma unroll
      for (int j = 0; j < 4; ++j) {
        const int dk = g + 32 * j;
        v16bf a = ld16bf(Wrow + dk + kbB, Wrow + dk + kbB + 8);
        c0 = __builtin_amdgcn_wmma_f32_16x16x32_bf16(false, a, false, bf0[j],
                                                     (short)0, c0, false, false);
        c1 = __builtin_amdgcn_wmma_f32_16x16x32_bf16(false, a, false, bf1[j],
                                                     (short)0, c1, false, false);
      }
    }
    const size_t base = ((size_t)b * cN + n0 + hf * 8) * cL + l0 + lq;
#pragma unroll
    for (int r = 0; r < 8; ++r) {
      const int n = n0 + hf * 8 + r;
      if (n < cN) {
        gA[base + (size_t)r * cL]      = c0[r];   // raw logits -> A region
        gA[base + (size_t)r * cL + 16] = c1[r];
      }
      rmax[r] = fmaxf(rmax[r], fmaxf(c0[r], c1[r]));  // per-lane partial max
    }
  }
  // cross-lane butterfly once per phase (columns live in 16-lane halves)
#pragma unroll
  for (int r = 0; r < 8; ++r) {
    float v = rmax[r];
    v = fmaxf(v, __shfl_xor(v, 1, 32));
    v = fmaxf(v, __shfl_xor(v, 2, 32));
    v = fmaxf(v, __shfl_xor(v, 4, 32));
    v = fmaxf(v, __shfl_xor(v, 8, 32));
    rmax[r] = v;
  }
  if (lq == 0) {
#pragma unroll
    for (int r = 0; r < 8; ++r) sPart[wv][hf * 8 + r] = rmax[r];
  }
  __threadfence_block();
  __syncthreads();
  if (tid < 16)
    sMax[tid] = fmaxf(fmaxf(sPart[0][tid], sPart[1][tid]),
                      fmaxf(sPart[2][tid], sPart[3][tid]));
  __syncthreads();

  // ---------------- Phase 2: p = exp(E - max), row sum ----------------
  float mx[8], rsum[8];
#pragma unroll
  for (int r = 0; r < 8; ++r) { mx[r] = sMax[hf * 8 + r]; rsum[r] = 0.f; }

  for (int blk = wv; blk < cL / 16; blk += 4) {
    const int l0 = blk * 16;
    const size_t base = ((size_t)b * cN + n0 + hf * 8) * cL + l0 + lq;
#pragma unroll
    for (int r = 0; r < 8; ++r) {
      const int n = n0 + hf * 8 + r;
      if (n < cN) {
        const size_t idx = base + (size_t)r * cL;
        const float p = __expf(gA[idx] - mx[r]);
        gA[idx] = p;
        rsum[r] += p;                               // per-lane partial sum
      }
    }
  }
#pragma unroll
  for (int r = 0; r < 8; ++r) {
    float v = rsum[r];
    v += __shfl_xor(v, 1, 32);
    v += __shfl_xor(v, 2, 32);
    v += __shfl_xor(v, 4, 32);
    v += __shfl_xor(v, 8, 32);
    rsum[r] = v;
  }
  if (lq == 0) {
#pragma unroll
    for (int r = 0; r < 8; ++r) sPart[wv][hf * 8 + r] = rsum[r];
  }
  __threadfence_block();
  __syncthreads();
  if (tid < 16) {
    const float s = sPart[0][tid] + sPart[1][tid] + sPart[2][tid] + sPart[3][tid];
    sInv[tid] = (s > 0.f) ? (1.0f / s) : 0.f;
  }
  __syncthreads();

  // ---------------- Phase 3: finalize A = p / sum ----------------
  float inv[8];
#pragma unroll
  for (int r = 0; r < 8; ++r) inv[r] = sInv[hf * 8 + r];

  for (int blk = wv; blk < cL / 16; blk += 4) {
    const int l0 = blk * 16;
    const size_t base = ((size_t)b * cN + n0 + hf * 8) * cL + l0 + lq;
#pragma unroll
    for (int r = 0; r < 8; ++r) {
      const int n = n0 + hf * 8 + r;
      if (n < cN) gA[base + (size_t)r * cL] *= inv[r];
    }
  }
  __threadfence_block();
  __syncthreads();

  // ---------------- Phase 4: C = A @ V^T (waves split D) ----------------
  const int d0w = wv * 128;                    // this wave's 128-wide d slice
  v8f acc[8];
#pragma unroll
  for (int s = 0; s < 8; ++s)
    acc[s] = (v8f){0.f, 0.f, 0.f, 0.f, 0.f, 0.f, 0.f, 0.f};

  const int  na = n0 + lq;                     // A-frag row
  const bool av = (na < cN);
  const float* Arow = gA + ((size_t)b * cN + (av ? na : n0)) * cL;
  const __bf16* Vbase = Vbf + ((size_t)b * cD + d0w + lq) * cL + kbB;

  for (int l0 = 0; l0 < cL; l0 += 32) {
    // A fragment: one contiguous 16-wide fp32 run per lane, converted to bf16
    const float4 a0 = *(const float4*)(Arow + l0 + kbB);
    const float4 a1 = *(const float4*)(Arow + l0 + kbB + 4);
    const float4 a2 = *(const float4*)(Arow + l0 + kbB + 8);
    const float4 a3 = *(const float4*)(Arow + l0 + kbB + 12);
    // 8 independent B fragments (one per d-subtile) -> 16-load clause
    v16bf bb[8];
#pragma unroll
    for (int s = 0; s < 8; ++s)
      bb[s] = ld16bf(Vbase + l0 + (size_t)(s * 16) * cL,
                     Vbase + l0 + (size_t)(s * 16) * cL + 8);
    v16bf a;
    a[0]  = (__bf16)a0.x; a[1]  = (__bf16)a0.y; a[2]  = (__bf16)a0.z; a[3]  = (__bf16)a0.w;
    a[4]  = (__bf16)a1.x; a[5]  = (__bf16)a1.y; a[6]  = (__bf16)a1.z; a[7]  = (__bf16)a1.w;
    a[8]  = (__bf16)a2.x; a[9]  = (__bf16)a2.y; a[10] = (__bf16)a2.z; a[11] = (__bf16)a2.w;
    a[12] = (__bf16)a3.x; a[13] = (__bf16)a3.y; a[14] = (__bf16)a3.z; a[15] = (__bf16)a3.w;
    if (!av) {
#pragma unroll
      for (int e = 0; e < 16; ++e) a[e] = (__bf16)0.0f;
    }
#pragma unroll
    for (int s = 0; s < 8; ++s)
      acc[s] = __builtin_amdgcn_wmma_f32_16x16x32_bf16(false, a, false, bb[s],
                                                       (short)0, acc[s],
                                                       false, false);
  }
#pragma unroll
  for (int s = 0; s < 8; ++s) {
#pragma unroll
    for (int r = 0; r < 8; ++r) {
      const int n = n0 + hf * 8 + r;
      if (n < cN)
        gC[((size_t)b * cN + n) * cD + d0w + s * 16 + lq] = acc[s][r];
    }
  }
}

// ============================ launcher ================================
extern "C" void kernel_launch(void* const* d_in, const int* in_sizes, int n_in,
                              void* d_out, int out_size, void* d_ws,
                              size_t ws_size, hipStream_t stream) {
  (void)in_sizes; (void)n_in; (void)out_size; (void)ws_size;

  const float* K = (const float*)d_in[0];  // [B,D,L]
  const float* V = (const float*)d_in[1];  // [B,D,L]
  const float* W = (const float*)d_in[2];  // [N,D]

  float* gC = (float*)d_out;                                   // [B,N,D]
  float* gA = (float*)d_out + (size_t)cB * cN * cD;            // [B,N,L]

  // workspace: bf16 staged operands (all L2-resident, ~43 MB total)
  char* ws = (char*)d_ws;
  size_t off = 0;
  __bf16* Wbf = (__bf16*)(ws + off); off += (size_t)cNpad * cD * 2;      // 9.1 MB
  __bf16* Kt  = (__bf16*)(ws + off); off += (size_t)cB * cL * cD * 2;    // 16.8 MB
  __bf16* Vbf = (__bf16*)(ws + off);                                     // 16.8 MB

  {
    const size_t nElem = (size_t)cNpad * cD;
    ta_prep_w<<<(unsigned)((nElem + 255) / 256), 256, 0, stream>>>(W, Wbf);
  }
  {
    const size_t nElem = (size_t)cB * cD * cL;
    ta_prep_v<<<(unsigned)((nElem + 255) / 256), 256, 0, stream>>>(V, Vbf);
  }
  {
    dim3 g(cL / 32, cD / 32, cB);
    dim3 t(32, 8);
    ta_prep_kt<<<g, t, 0, stream>>>(K, Kt);
  }
  ta_attn<<<cB * cNT, 128, 0, stream>>>(Wbf, Kt, Vbf, gA, gC);
}
